// CrossAttention_35304631173270
// MI455X (gfx1250) — matrix operand order repaired and checked by reading, not compiled, not run on hardware
//
#include <hip/hip_runtime.h>
#include <hip/hip_bf16.h>

typedef __bf16 bf16;
typedef __attribute__((ext_vector_type(16))) __bf16 v16bf;
typedef __attribute__((ext_vector_type(8)))  __bf16 v8bf;
typedef __attribute__((ext_vector_type(8)))  float  v8f;

#define WMMA_BF16(a, b, c) \
  __builtin_amdgcn_wmma_f32_16x16x32_bf16(false, (a), false, (b), (short)0, (c), false, false)

__device__ __forceinline__ v8f vzero8() {
  v8f z = {0.f, 0.f, 0.f, 0.f, 0.f, 0.f, 0.f, 0.f};
  return z;
}

// A fragment (16x32 bf16, row-major source): lane l holds row l%16.
// elements 0..7  = K[k0 + 16*(l/16) .. +7]       (contiguous 16B)
// elements 8..15 = K[k0 + 16*(l/16) + 8 .. +7]   (contiguous 16B)
__device__ __forceinline__ v16bf load_a_frag(const bf16* A, int lda, int row0, int k0) {
  const int lane = threadIdx.x & 31;
  const int half = lane >> 4;
  const bf16* p = A + (size_t)(row0 + (lane & 15)) * lda + (k0 + half * 16);
  v8bf lo = *(const v8bf*)p;
  v8bf hi = *(const v8bf*)(p + 8);
  v16bf r;
#pragma unroll
  for (int i = 0; i < 8; ++i) { r[i] = lo[i]; r[i + 8] = hi[i]; }
  return r;
}

// B fragment (32x16 bf16) loaded from B^T stored row-major [N,K]:
// lane l holds col n0 + l%16; elements e=0..15 are K[k0 + 16*(l/16) + e] (contiguous 32B)
__device__ __forceinline__ v16bf load_b_frag(const bf16* Bt, int ldb, int n0, int k0) {
  const int lane = threadIdx.x & 31;
  const int half = lane >> 4;
  const bf16* p = Bt + (size_t)(n0 + (lane & 15)) * ldb + (k0 + half * 16);
  return *(const v16bf*)p;
}

// Load one full fragment set (32xK-step A pair + 64-wide B quad) for K offset k0.
__device__ __forceinline__ void load_frag_set(const bf16* A, int lda, const bf16* Bt, int ldb,
                                              int row0, int col0, int k0,
                                              v16bf& a0, v16bf& a1, v16bf b[4]) {
  a0 = load_a_frag(A, lda, row0, k0);
  a1 = load_a_frag(A, lda, row0 + 16, k0);
  b[0] = load_b_frag(Bt, ldb, col0 +  0, k0);
  b[1] = load_b_frag(Bt, ldb, col0 + 16, k0);
  b[2] = load_b_frag(Bt, ldb, col0 + 32, k0);
  b[3] = load_b_frag(Bt, ldb, col0 + 48, k0);
}

__device__ __forceinline__ void wmma8(const v16bf& a0, const v16bf& a1, const v16bf b[4],
                                      v8f acc[8]) {
  acc[0] = WMMA_BF16(a0, b[0], acc[0]);
  acc[1] = WMMA_BF16(a0, b[1], acc[1]);
  acc[2] = WMMA_BF16(a0, b[2], acc[2]);
  acc[3] = WMMA_BF16(a0, b[3], acc[3]);
  acc[4] = WMMA_BF16(a1, b[0], acc[4]);
  acc[5] = WMMA_BF16(a1, b[1], acc[5]);
  acc[6] = WMMA_BF16(a1, b[2], acc[6]);
  acc[7] = WMMA_BF16(a1, b[3], acc[7]);
}

// One wave computes a 32(M) x 64(N) f32 tile. Ping-pong pipelined K loop, step 64
// (two fragment sets, no register rotation copies). Requires K % 64 == 0.
__device__ __forceinline__ void gemm_tile_32x64(const bf16* A, int lda, const bf16* Bt, int ldb,
                                                int row0, int col0, int K, v8f acc[8]) {
#pragma unroll
  for (int j = 0; j < 8; ++j) acc[j] = vzero8();

  v16bf a0A, a1A, bA[4], a0B, a1B, bB[4];
  load_frag_set(A, lda, Bt, ldb, row0, col0, 0, a0A, a1A, bA);

  for (int k0 = 0; k0 < K; k0 += 64) {
    load_frag_set(A, lda, Bt, ldb, row0, col0, k0 + 32, a0B, a1B, bB);
    __builtin_prefetch((const void*)(A + (size_t)row0 * lda + k0 + 64), 0, 3);
    wmma8(a0A, a1A, bA, acc);
    if (k0 + 64 < K)
      load_frag_set(A, lda, Bt, ldb, row0, col0, k0 + 64, a0A, a1A, bA);
    wmma8(a0B, a1B, bB, acc);
  }
}

// ---------------- conversion kernels ----------------
__global__ void k_cvt(const float* __restrict__ in, bf16* __restrict__ out, int n) {
  int i = blockIdx.x * 256 + threadIdx.x;
  if (i < n) out[i] = (bf16)in[i];
}

// in [rows][cols] fp32 -> out [cols][rows] bf16 (weight transpose)
__global__ void k_cvt_t(const float* __restrict__ in, bf16* __restrict__ out, int rows, int cols) {
  int i = blockIdx.x * 256 + threadIdx.x;
  if (i < rows * cols) {
    int r = i / cols, c = i - r * cols;
    out[(size_t)c * rows + r] = (bf16)in[i];
  }
}

// ---------------- KV projection: [4096,1024] @ Wkv -> K cache + V^T cache ----------------
__global__ void __launch_bounds__(256, 1)
k_gemm_kv(const bf16* __restrict__ memb, const bf16* __restrict__ wkvt,
          bf16* __restrict__ kc, bf16* __restrict__ vt) {
  int widx = (blockIdx.x * blockDim.x + threadIdx.x) >> 5;
  int lane = threadIdx.x & 31, half = lane >> 4, ln = lane & 15;
  int mtile = widx & 127;      // 128 row tiles of 32 (4096 rows)
  int ntile = widx >> 7;       // 32 col tiles of 64 (2048 cols)
  int row0 = mtile * 32, col0 = ntile * 64;
  v8f acc[8];
  gemm_tile_32x64(memb, 1024, wkvt, 1024, row0, col0, 1024, acc);
#pragma unroll
  for (int hh = 0; hh < 2; ++hh)
#pragma unroll
    for (int j = 0; j < 4; ++j) {
      int col = col0 + 16 * j + ln;
#pragma unroll
      for (int r = 0; r < 8; ++r) {
        int row = row0 + 16 * hh + 8 * half + r;   // global row = b*2048 + kv
        int b = row >> 11, kv = row & 2047;
        bf16 val = (bf16)acc[hh * 4 + j][r];
        if (col < 1024) {                          // K head: [b,h,kv,hd]
          int h = col >> 6, d = col & 63;
          kc[((size_t)((b * 16 + h) * 2048) + kv) * 64 + d] = val;
        } else {                                   // V head, transposed: [b,h,hd,kv]
          int c2 = col - 1024, h = c2 >> 6, d = c2 & 63;
          vt[((size_t)((b * 16 + h) * 64) + d) * 2048 + kv] = val;
        }
      }
    }
}

// ---------------- Q projection: [2048,1024] @ Wq -> Q cache [b,h,q,hd] ----------------
__global__ void __launch_bounds__(256, 1)
k_gemm_q(const bf16* __restrict__ xb, const bf16* __restrict__ wqt,
         bf16* __restrict__ qc) {
  int widx = (blockIdx.x * blockDim.x + threadIdx.x) >> 5;
  int lane = threadIdx.x & 31, half = lane >> 4, ln = lane & 15;
  int mtile = widx & 63;       // 64 row tiles of 32 (2048 rows)
  int ntile = widx >> 6;       // 16 col tiles of 64 (1024 cols)
  int row0 = mtile * 32, col0 = ntile * 64;
  v8f acc[8];
  gemm_tile_32x64(xb, 1024, wqt, 1024, row0, col0, 1024, acc);
#pragma unroll
  for (int hh = 0; hh < 2; ++hh)
#pragma unroll
    for (int j = 0; j < 4; ++j) {
      int col = col0 + 16 * j + ln;
      int h = col >> 6, d = col & 63;
#pragma unroll
      for (int r = 0; r < 8; ++r) {
        int row = row0 + 16 * hh + 8 * half + r;   // global row = b*1024 + q
        int b = row >> 10, q = row & 1023;
        qc[((size_t)((b * 16 + h) * 1024) + q) * 64 + d] = (bf16)acc[hh * 4 + j][r];
      }
    }
}

// ---------------- flash attention: one wave per (b,h,32-row q tile) ----------------
__global__ void __launch_bounds__(256, 1)
k_attn(const bf16* __restrict__ qc, const bf16* __restrict__ kc,
       const bf16* __restrict__ vt, bf16* __restrict__ o) {
  __shared__ __align__(32) bf16 pbuf[8 * 32 * 32];   // per-wave 32x32 P staging
  int wave = threadIdx.x >> 5;
  int widx = (blockIdx.x * blockDim.x + threadIdx.x) >> 5;
  int lane = threadIdx.x & 31, half = lane >> 4, ln = lane & 15;
  int qt = widx & 31;              // 32 q tiles of 32
  int h  = (widx >> 5) & 15;
  int b  = widx >> 9;

  const bf16* qptr = qc + (size_t)((b * 16 + h) * 1024 + qt * 32) * 64; // [32 x 64]
  const bf16* kptr = kc + (size_t)(b * 16 + h) * 2048 * 64;            // [kv x 64]
  const bf16* vptr = vt + (size_t)(b * 16 + h) * 64 * 2048;            // [64 x kv]
  bf16* pw = pbuf + wave * (32 * 32);

  // Q fragments: qa[qh][kk] -> rows [16qh,+16), K chunk kk*32
  v16bf qa[2][2];
#pragma unroll
  for (int qh = 0; qh < 2; ++qh) {
    qa[qh][0] = load_a_frag(qptr, 64, 16 * qh, 0);
    qa[qh][1] = load_a_frag(qptr, 64, 16 * qh, 32);
  }

  float m[2][8], l[2][8];
#pragma unroll
  for (int qh = 0; qh < 2; ++qh)
#pragma unroll
    for (int r = 0; r < 8; ++r) { m[qh][r] = -1e30f; l[qh][r] = 0.f; }
  v8f oacc[2][4];
#pragma unroll
  for (int qh = 0; qh < 2; ++qh)
#pragma unroll
    for (int j = 0; j < 4; ++j) oacc[qh][j] = vzero8();

  const float scale = 0.125f;      // 64^-0.5

  for (int kv0 = 0; kv0 < 2048; kv0 += 32) {
    // K fragments for the two 16-wide kv tiles (reused by both q halves)
    v16bf kb[2][2];
#pragma unroll
    for (int t = 0; t < 2; ++t) {
      kb[t][0] = load_b_frag(kptr, 64, kv0 + 16 * t, 0);
      kb[t][1] = load_b_frag(kptr, 64, kv0 + 16 * t, 32);
    }
    // S = q @ k^T
    v8f s[2][2];
#pragma unroll
    for (int qh = 0; qh < 2; ++qh)
#pragma unroll
      for (int t = 0; t < 2; ++t) {
        v8f acc = vzero8();
        acc = WMMA_BF16(qa[qh][0], kb[t][0], acc);
        acc = WMMA_BF16(qa[qh][1], kb[t][1], acc);
        s[qh][t] = acc;
      }

    // issue V fragments now so their latency hides under softmax VALU work
    v16bf vb[4];
#pragma unroll
    for (int j = 0; j < 4; ++j) vb[j] = load_b_frag(vptr, 2048, 16 * j, kv0);

    // online softmax (element r of q-half qh == q row 16qh + 8*half + r)
    float alpha[2][8];
#pragma unroll
    for (int qh = 0; qh < 2; ++qh)
#pragma unroll
      for (int r = 0; r < 8; ++r) {
        float a = s[qh][0][r] * scale, bv = s[qh][1][r] * scale;
        float mx = fmaxf(a, bv);
#pragma unroll
        for (int off = 1; off < 16; off <<= 1) mx = fmaxf(mx, __shfl_xor(mx, off, 32));
        float mn = fmaxf(m[qh][r], mx);
        alpha[qh][r] = __expf(m[qh][r] - mn);
        m[qh][r] = mn;
        float p0 = __expf(a - mn), p1 = __expf(bv - mn);
        float rs = p0 + p1;
#pragma unroll
        for (int off = 1; off < 16; off <<= 1) rs += __shfl_xor(rs, off, 32);
        l[qh][r] = l[qh][r] * alpha[qh][r] + rs;
        int row = 16 * qh + 8 * half + r;          // C-layout -> row-major P in LDS
        pw[row * 32 + ln]      = (bf16)p0;
        pw[row * 32 + 16 + ln] = (bf16)p1;
      }
    __syncthreads();
    v16bf pa0 = load_a_frag(pw, 32, 0, 0);         // P as two 16x32 A fragments
    v16bf pa1 = load_a_frag(pw, 32, 16, 0);
    __syncthreads();
#pragma unroll
    for (int j = 0; j < 4; ++j) {
#pragma unroll
      for (int r = 0; r < 8; ++r) { oacc[0][j][r] *= alpha[0][r]; oacc[1][j][r] *= alpha[1][r]; }
      oacc[0][j] = WMMA_BF16(pa0, vb[j], oacc[0][j]);
      oacc[1][j] = WMMA_BF16(pa1, vb[j], oacc[1][j]);
    }
  }

#pragma unroll
  for (int qh = 0; qh < 2; ++qh)
#pragma unroll
    for (int j = 0; j < 4; ++j)
#pragma unroll
      for (int r = 0; r < 8; ++r) {
        int q = qt * 32 + 16 * qh + 8 * half + r;
        float v = oacc[qh][j][r] / l[qh][r];
        o[(size_t)(b * 1024 + q) * 1024 + h * 64 + 16 * j + ln] = (bf16)v;
      }
}

// ---------------- output projection + bias: [2048,1024] @ Wproj + b -> fp32 ----------------
__global__ void __launch_bounds__(256, 1)
k_gemm_proj(const bf16* __restrict__ ob, const bf16* __restrict__ wpt,
            const float* __restrict__ bias, float* __restrict__ out) {
  int widx = (blockIdx.x * blockDim.x + threadIdx.x) >> 5;
  int lane = threadIdx.x & 31, half = lane >> 4, ln = lane & 15;
  int mtile = widx & 63;       // 2048 rows / 32
  int ntile = widx >> 6;       // 1024 cols / 64
  int row0 = mtile * 32, col0 = ntile * 64;
  v8f acc[8];
  gemm_tile_32x64(ob, 1024, wpt, 1024, row0, col0, 1024, acc);
#pragma unroll
  for (int hh = 0; hh < 2; ++hh)
#pragma unroll
    for (int j = 0; j < 4; ++j) {
      int col = col0 + 16 * j + ln;
      float bb = bias[col];
#pragma unroll
      for (int r = 0; r < 8; ++r) {
        int row = row0 + 16 * hh + 8 * half + r;
        out[(size_t)row * 1024 + col] = acc[hh * 4 + j][r] + bb;
      }
    }
}

extern "C" void kernel_launch(void* const* d_in, const int* in_sizes, int n_in,
                              void* d_out, int out_size, void* d_ws, size_t ws_size,
                              hipStream_t stream) {
  const float* x      = (const float*)d_in[0];  // [2,1024,1024]
  const float* mem    = (const float*)d_in[1];  // [2,2048,1024]
  const float* W_kv   = (const float*)d_in[2];  // [1024,2048]
  const float* W_q    = (const float*)d_in[3];  // [1024,1024]
  const float* W_proj = (const float*)d_in[4];  // [1024,1024]
  const float* b_proj = (const float*)d_in[5];  // [1024]
  float* out = (float*)d_out;

  const size_t M1 = 1u << 20;
  bf16* ws   = (bf16*)d_ws;
  bf16* xb   = ws;             // 2M
  bf16* memb = ws + 2 * M1;    // 4M
  bf16* wkvt = ws + 6 * M1;    // 2M  (W_kv^T [2048,1024])
  bf16* wqt  = ws + 8 * M1;    // 1M  (W_q^T  [1024,1024])
  bf16* wpt  = ws + 9 * M1;    // 1M  (W_proj^T)
  bf16* kc   = ws + 10 * M1;   // 4M  K cache [b,h,kv,64]
  bf16* vt   = ws + 14 * M1;   // 4M  V^T cache [b,h,64,kv]
  bf16* qc   = ws + 18 * M1;   // 2M  Q cache [b,h,q,64]
  bf16* ob   = ws + 20 * M1;   // 2M  attn out [b*q, 1024]

  // stage 1: precision conversion (+ weight transposes)
  k_cvt<<<(2 * M1) / 256, 256, 0, stream>>>(x, xb, 2 * M1);
  k_cvt<<<(4 * M1) / 256, 256, 0, stream>>>(mem, memb, 4 * M1);
  k_cvt_t<<<(2 * M1) / 256, 256, 0, stream>>>(W_kv, wkvt, 1024, 2048);
  k_cvt_t<<<(1 * M1) / 256, 256, 0, stream>>>(W_q, wqt, 1024, 1024);
  k_cvt_t<<<(1 * M1) / 256, 256, 0, stream>>>(W_proj, wpt, 1024, 1024);

  // stage 2: KV projection (4096 waves = 512 blocks of 8 waves)
  k_gemm_kv<<<512, 256, 0, stream>>>(memb, wkvt, kc, vt);
  // stage 3: Q projection (1024 waves)
  k_gemm_q<<<128, 256, 0, stream>>>(xb, wqt, qc);
  // stage 4: flash attention (1024 waves: b*h*qtiles = 2*16*32)
  k_attn<<<128, 256, 0, stream>>>(qc, kc, vt, ob);
  // stage 5: output projection + bias
  k_gemm_proj<<<128, 256, 0, stream>>>(ob, wpt, b_proj, out);
}
